// MetScore_2164663517874
// MI455X (gfx1250) — compile-verified, matrix operand order checked by reading, not run on hardware
//
#include <hip/hip_runtime.h>
#include <cmath>

// ---------------------------------------------------------------------------
// MetScore for B=8, T=20, H=512, W=512 fp32 inputs.
// Memory-bound: 503 MB streamed once -> ~21.6 us floor at 23.3 TB/s.
// Stage 1: streaming per-block partial reduction (NT b128 loads, wave32
//          shuffle + LDS reduce). Stage 2: deterministic cross-block reduce.
// Stage 3: epilogue; 20x{moments,level stats} -> 251 outputs, using one
//          v_wmma_f32_16x16x32_f16 (B = ones) to do all row-sums over time.
// ---------------------------------------------------------------------------

typedef float     v4f  __attribute__((ext_vector_type(4)));
typedef _Float16  v16h __attribute__((ext_vector_type(16)));
typedef float     v8f  __attribute__((ext_vector_type(8)));

#define HWSZ      262144   // 512*512 elements per (b,t) slab
#define TSTEPS    20
#define NLEV      5
#define NACC      26       // 6 moments + 5*(hits,miss,fas,errsum)
#define ACC_STRIDE 32
#define CHUNKS    32       // blocks per slab
#define C4        2048     // float4 per block  (HWSZ/4/CHUNKS)

__constant__ float c_tw[TSTEPS] = {
    0.0075f,0.02f,0.03f,0.04f,0.05f,0.06f,0.07f,0.08f,0.09f,0.1f,
    0.09f,0.08f,0.07f,0.06f,0.05f,0.04f,0.03f,0.02f,0.0075f,0.005f};

__global__ void metscore_zero(float* __restrict__ acc) {
    for (int i = threadIdx.x; i < TSTEPS * ACC_STRIDE; i += blockDim.x)
        acc[i] = 0.f;
}

// ---- Stage 1: per-block partials --------------------------------------------
__global__ __launch_bounds__(256)
void metscore_partial(const float* __restrict__ pred,
                      const float* __restrict__ targ,
                      const float* __restrict__ mask,
                      float* __restrict__ acc,       // atomic fallback target
                      float* __restrict__ dpart)     // deterministic partials (or null)
{
    const int    slab  = blockIdx.x;           // slab = b*T + t
    const int    tstep = slab % TSTEPS;
    const size_t base  = (size_t)slab * HWSZ;
    const v4f* p4 = (const v4f*)(pred + base);
    const v4f* t4 = (const v4f*)(targ + base);
    const v4f* m4 = (const v4f*)(mask + base);
    const int  i0 = blockIdx.y * C4 + threadIdx.x;

    float A[NACC];
#pragma unroll
    for (int i = 0; i < NACC; ++i) A[i] = 0.f;

#pragma unroll
    for (int bat = 0; bat < 2; ++bat) {
        v4f P[4], Q[4], M[4];
#pragma unroll
        for (int k = 0; k < 4; ++k) {
            const int idx = i0 + (bat * 4 + k) * 256;
            P[k] = __builtin_nontemporal_load(p4 + idx);   // global_load_b128 th:NT
            Q[k] = __builtin_nontemporal_load(t4 + idx);
            M[k] = __builtin_nontemporal_load(m4 + idx);
        }
#pragma unroll
        for (int k = 0; k < 4; ++k) {
#pragma unroll
            for (int j = 0; j < 4; ++j) {
                const float p   = P[k][j] * 30.f;          // DATA_MAX
                const float t   = Q[k][j] * 30.f;
                const bool  msk = M[k][j] > 0.5f;
                const bool  dz  = (p < 0.1f) && (t < 0.1f);
                const float mc  = (msk && !dz) ? 1.f : 0.f;
                A[0] += mc;
                A[1] += p * mc;
                A[2] += t * mc;
                A[3] += p * p * mc;
                A[4] += t * t * mc;
                A[5] += p * t * mc;
                const int bp = (p >= 0.1f) + (p >= 1.f) + (p >= 2.f) + (p >= 5.f) + (p >= 8.f);
                const int bt = (t >= 0.1f) + (t >= 1.f) + (t >= 2.f) + (t >= 5.f) + (t >= 8.f);
                const float ae = fabsf(p - t);
#pragma unroll
                for (int l = 0; l < NLEV; ++l) {
                    const bool ip = msk && (bp == l + 1);
                    const bool it = msk && (bt == l + 1);
                    A[6 + 4*l + 0] += (ip && it)  ? 1.f : 0.f;   // hits
                    A[6 + 4*l + 1] += (it && !ip) ? 1.f : 0.f;   // misses
                    A[6 + 4*l + 2] += (ip && !it) ? 1.f : 0.f;   // false alarms
                    A[6 + 4*l + 3] += it ? ae : 0.f;             // err sum
                }
            }
        }
    }

    // wave32 shuffle reduce, then combine 8 waves through LDS
    __shared__ float part[8][NACC];
    const int lane = threadIdx.x & 31;
    const int wave = threadIdx.x >> 5;
#pragma unroll
    for (int i = 0; i < NACC; ++i) {
        float v = A[i];
#pragma unroll
        for (int off = 16; off > 0; off >>= 1)
            v += __shfl_down(v, off, 32);
        if (lane == 0) part[wave][i] = v;
    }
    __syncthreads();
    if (threadIdx.x < NACC) {
        float s = 0.f;
#pragma unroll
        for (int w = 0; w < 8; ++w) s += part[w][threadIdx.x];
        if (dpart) {
            dpart[((size_t)slab * CHUNKS + blockIdx.y) * NACC + threadIdx.x] = s;
        } else {
            atomicAdd(&acc[tstep * ACC_STRIDE + threadIdx.x], s);
        }
    }
}

// ---- Stage 2: deterministic cross-block reduction ---------------------------
__global__ void metscore_reduce(const float* __restrict__ dpart,
                                float* __restrict__ acc, int slabs)
{
    const int t = blockIdx.x;
    const int k = threadIdx.x;
    if (k >= NACC) return;
    const int nb = slabs / TSTEPS;      // batches per timestep (8)
    float s = 0.f;
    for (int b = 0; b < nb; ++b)
        for (int c = 0; c < CHUNKS; ++c) {
            const size_t row = (size_t)(b * TSTEPS + t) * CHUNKS + c;
            s += dpart[row * NACC + k];
        }
    acc[t * ACC_STRIDE + k] = s;
}

// ---- Stage 3: epilogue (single wave; WMMA row-sums over time) ---------------
__global__ void metscore_final(const float* __restrict__ acc,
                               float* __restrict__ out)
{
    __shared__ float S[16][32];         // A-matrix staging: rows x time
    const int lane = threadIdx.x;       // 0..31, single wave
    for (int i = lane; i < 16 * 32; i += 32) ((float*)S)[i] = 0.f;
    __syncthreads();

    const float LW[NLEV] = {0.1f, 0.1f, 0.2f, 0.25f, 0.35f};

    if (lane < TSTEPS) {
        const float* a = acc + lane * ACC_STRIDE;
        const float n = a[0], sp = a[1], st = a[2], spp = a[3], stt = a[4], spt = a[5];
        const float ns = fmaxf(n, 1.f);
        const float pm = sp / ns, tm = st / ns;
        const float num = spt - pm * st - tm * sp + pm * tm * n;
        const float vp  = fmaxf(spp - 2.f * pm * sp + pm * pm * n, 0.f);
        const float vt  = fmaxf(stt - 2.f * tm * st + tm * tm * n, 0.f);
        const float den = sqrtf(vp * vt);
        const float R   = fminf(fmaxf(num / (den + 1e-6f), -1.f), 1.f);
        const float r   = (n > 0.f) ? R : 0.f;
        const float term_corr = sqrtf(__expf(r - 1.f));
        float sl = 0.f;
#pragma unroll
        for (int l = 0; l < NLEV; ++l) {
            const float h  = a[6 + 4*l], mi = a[7 + 4*l];
            const float fa = a[8 + 4*l], es = a[9 + 4*l];
            const float ts = h / (h + mi + fa + 1e-8f);
            const float nb = h + mi;
            const float mae = (nb > 0.f) ? es / fmaxf(nb, 1.f) : 0.f;
            const float tmae = sqrtf(__expf(-mae * 0.01f));
            sl += LW[l] * ts * tmae;
            out[41  + lane * NLEV + l] = ts;    // ts_time  [20,5]
            out[141 + lane * NLEV + l] = mae;   // mae_time [20,5]
            S[l][lane]     = ts;                // rows 0..4  : ts per level
            S[5 + l][lane] = mae;               // rows 5..9  : mae per level
        }
        const float score = term_corr * sl;
        out[1  + lane] = score;                 // score_time [20]
        out[21 + lane] = r;                     // r_time     [20]
        S[10][lane] = score * c_tw[lane];       // row 10     : weighted score
    }
    __syncthreads();

    // A (16x32 f16) per CDNA5 layout: lanes 0-15 hold K=0..7,16..23 of row m,
    // lanes 16-31 hold K=8..15,24..31.  B = all-ones (layout-invariant).
    const int m  = lane & 15;
    const int ko = (lane >> 4) ? 8 : 0;
    v16h av, bv;
#pragma unroll
    for (int e = 0; e < 8; ++e) av[e]     = (_Float16)S[m][ko + e];
#pragma unroll
    for (int e = 0; e < 8; ++e) av[8 + e] = (_Float16)S[m][16 + ko + e];
#pragma unroll
    for (int e = 0; e < 16; ++e) bv[e] = (_Float16)1.0f;
    v8f cv = {};
    v8f d = __builtin_amdgcn_wmma_f32_16x16x32_f16(false, av, false, bv,
                                                   (short)0, cv, false, false);
    // D layout: lanes 0-15: d[v] = rowsum(M=v); lanes 16-31: d[v] = rowsum(M=v+8)
    const float inv20 = 1.f / 20.f;
    if (lane == 0) {
        out[241] = d[0] * inv20; out[242] = d[1] * inv20; out[243] = d[2] * inv20;
        out[244] = d[3] * inv20; out[245] = d[4] * inv20;                    // ts_levels
        out[246] = d[5] * inv20; out[247] = d[6] * inv20; out[248] = d[7] * inv20; // mae_levels[0..2]
    }
    if (lane == 16) {
        out[249] = d[0] * inv20; out[250] = d[1] * inv20;   // mae_levels[3..4]
        out[0]   = d[2];                                    // total_score (row 10)
    }
}

// ---------------------------------------------------------------------------
extern "C" void kernel_launch(void* const* d_in, const int* in_sizes, int n_in,
                              void* d_out, int out_size, void* d_ws, size_t ws_size,
                              hipStream_t stream) {
    const float* pred = (const float*)d_in[0];
    const float* targ = (const float*)d_in[1];
    const float* mask = (const float*)d_in[2];
    float* out = (float*)d_out;

    const int slabs = in_sizes[0] / HWSZ;                 // B*T = 160
    const size_t acc_bytes  = (size_t)TSTEPS * ACC_STRIDE * sizeof(float);
    const size_t part_bytes = (size_t)slabs * CHUNKS * NACC * sizeof(float);

    float* accbuf  = (float*)d_ws;
    float* partbuf = (float*)((char*)d_ws + acc_bytes);
    const bool det = (ws_size >= acc_bytes + part_bytes); // deterministic path if ws fits

    metscore_zero<<<1, 256, 0, stream>>>(accbuf);
    metscore_partial<<<dim3(slabs, CHUNKS), 256, 0, stream>>>(
        pred, targ, mask, accbuf, det ? partbuf : nullptr);
    if (det)
        metscore_reduce<<<TSTEPS, 32, 0, stream>>>(partbuf, accbuf, slabs);
    metscore_final<<<1, 32, 0, stream>>>(accbuf, out);
}
